// CoreferenceResolver_51539607552866
// MI455X (gfx1250) — matrix operand order repaired and checked by reading, not compiled, not run on hardware
//
#include <hip/hip_runtime.h>
#include <math.h>

// ---------------- problem constants ----------------
static constexpr int CB   = 4;     // batch
static constexpr int CL   = 1024;  // seq len
static constexpr int CD   = 768;   // model dim
static constexpr int CNE  = 32;    // entities
static constexpr int CP   = 992;   // pairs per batch
static constexpr int CNPAIR = CB * CP;  // 3968 = 248*16
static constexpr int CK   = 1024;  // D + OUT_CH
static constexpr int COC  = 256;   // OUT_CH

// ---------------- WMMA types ----------------
typedef __attribute__((ext_vector_type(16))) __bf16 v16bf;
typedef __attribute__((ext_vector_type(8)))  float  v8f;
typedef __attribute__((address_space(3))) unsigned short lds_us;

union FragBF { v16bf v; uint4 u[2]; };

__device__ __forceinline__ unsigned short f2bf(float f) {
  union { float f; unsigned int u; } c; c.f = f;
  unsigned int u = c.u + (0x7fffu + ((c.u >> 16) & 1u));  // round-to-nearest-even
  return (unsigned short)(u >> 16);
}

// ---------------- 1) entity gather + normalize ----------------
__global__ void k_gather_norm(const float* __restrict__ x,
                              const int* __restrict__ epos,
                              float* __restrict__ ent,
                              float* __restrict__ nrm) {
  int e = blockIdx.x;                  // b*CNE + n
  int tid = threadIdx.x;               // 256 threads
  int start = epos[e * 2];
  int idx = start + 1; if (idx > CL - 1) idx = CL - 1;
  bool valid = (start + 1) < CL;       // c == L in reference
  int b = e >> 5;
  const float* src = x + ((size_t)b * CL + idx) * CD;
  float v0 = valid ? src[tid]       : 0.f;
  float v1 = valid ? src[tid + 256] : 0.f;
  float v2 = valid ? src[tid + 512] : 0.f;
  __shared__ float red[256];
  red[tid] = v0 * v0 + v1 * v1 + v2 * v2;
  __syncthreads();
  for (int s = 128; s > 0; s >>= 1) {
    if (tid < s) red[tid] += red[tid + s];
    __syncthreads();
  }
  float inv = 1.f / fmaxf(sqrtf(red[0]), 1e-13f);
  float* eo = ent + (size_t)e * CD;
  float* no = nrm + (size_t)e * CD;
  eo[tid] = v0; eo[tid + 256] = v1; eo[tid + 512] = v2;
  no[tid] = v0 * inv; no[tid + 256] = v1 * inv; no[tid + 512] = v2 * inv;
}

// ---------------- 2) cosine map [B,1,32,32] ----------------
__global__ void k_cos(const float* __restrict__ nrm, float* __restrict__ cosm) {
  int b = blockIdx.x;
  int m = threadIdx.x, n = threadIdx.y;     // (32,32)
  const float* rn = nrm + ((size_t)(b * CNE + n)) * CD;
  const float* rm = nrm + ((size_t)(b * CNE + m)) * CD;
  float acc = 0.f;
  for (int k = 0; k < CD; ++k) acc += rn[k] * rm[k];
  cosm[(size_t)b * CNE * CNE + n * CNE + m] = acc;
}

// ---------------- 3) UNet building blocks ----------------
// dual-input 3x3 conv (+bias, relu) : channels [0,C1) from in1, [C1,C1+C2) from in2
__global__ void k_conv3x3(const float* __restrict__ in1, int C1,
                          const float* __restrict__ in2, int C2,
                          const float* __restrict__ w,
                          const float* __restrict__ bias,
                          float* __restrict__ out,
                          int Cout, int HW) {
  int bc = blockIdx.x;
  int b = bc / Cout, co = bc % Cout;
  int t = threadIdx.x;
  int y = t / HW, xg = t % HW;
  int Cin = C1 + C2;
  float acc = bias[co];
  const float* wbase = w + (size_t)co * Cin * 9;
  for (int ci = 0; ci < Cin; ++ci) {
    const float* ip = (ci < C1)
        ? in1 + ((size_t)b * C1 + ci) * HW * HW
        : in2 + ((size_t)b * C2 + (ci - C1)) * HW * HW;
    const float* wp = wbase + (size_t)ci * 9;
#pragma unroll
    for (int ky = 0; ky < 3; ++ky) {
      int iy = y + ky - 1;
      if (iy < 0 || iy >= HW) continue;
#pragma unroll
      for (int kx = 0; kx < 3; ++kx) {
        int ix = xg + kx - 1;
        if (ix < 0 || ix >= HW) continue;
        acc += ip[iy * HW + ix] * wp[ky * 3 + kx];
      }
    }
  }
  out[((size_t)b * Cout + co) * HW * HW + t] = fmaxf(acc, 0.f);
}

__global__ void k_pool(const float* __restrict__ in, float* __restrict__ out, int HWi) {
  int plane = blockIdx.x;
  int HWo = HWi >> 1;
  int t = threadIdx.x;
  int y = t / HWo, x = t % HWo;
  const float* ip = in + (size_t)plane * HWi * HWi;
  float a = ip[(2 * y) * HWi + 2 * x];
  a = fmaxf(a, ip[(2 * y) * HWi + 2 * x + 1]);
  a = fmaxf(a, ip[(2 * y + 1) * HWi + 2 * x]);
  a = fmaxf(a, ip[(2 * y + 1) * HWi + 2 * x + 1]);
  out[(size_t)plane * HWo * HWo + t] = a;
}

__global__ void k_up2(const float* __restrict__ in, float* __restrict__ out, int HWo) {
  int plane = blockIdx.x;
  int HWi = HWo >> 1;
  int t = threadIdx.x;
  int y = t / HWo, x = t % HWo;
  out[(size_t)plane * HWo * HWo + t] =
      in[(size_t)plane * HWi * HWi + (y >> 1) * HWi + (x >> 1)];
}

// t = relu(Wg·g + Wx·s)   (1x1 convs)
__global__ void k_gate_t(const float* __restrict__ g, int Cg,
                         const float* __restrict__ s, int Cs,
                         const float* __restrict__ wg,
                         const float* __restrict__ wx,
                         float* __restrict__ t, int Ct, int HW) {
  int bc = blockIdx.x;
  int b = bc / Ct, co = bc % Ct;
  int pix = threadIdx.x;
  int npix = HW * HW;
  float acc = 0.f;
  const float* gp = g + (size_t)b * Cg * npix + pix;
  const float* wgp = wg + (size_t)co * Cg;
  for (int c = 0; c < Cg; ++c) acc += wgp[c] * gp[(size_t)c * npix];
  const float* sp = s + (size_t)b * Cs * npix + pix;
  const float* wxp = wx + (size_t)co * Cs;
  for (int c = 0; c < Cs; ++c) acc += wxp[c] * sp[(size_t)c * npix];
  t[((size_t)b * Ct + co) * npix + pix] = fmaxf(acc, 0.f);
}

// out = s * sigmoid(psi·t)   (Cs == Ct)
__global__ void k_gate_apply(const float* __restrict__ t, int Ct,
                             const float* __restrict__ psi,
                             const float* __restrict__ s,
                             float* __restrict__ out, int HW) {
  int b = blockIdx.x;
  int pix = threadIdx.x;
  int npix = HW * HW;
  float a = 0.f;
  const float* tp = t + (size_t)b * Ct * npix + pix;
  for (int c = 0; c < Ct; ++c) a += psi[c] * tp[(size_t)c * npix];
  a = 1.f / (1.f + expf(-a));
  const float* sp = s + (size_t)b * Ct * npix + pix;
  float* op = out + (size_t)b * Ct * npix + pix;
  for (int c = 0; c < Ct; ++c) op[(size_t)c * npix] = sp[(size_t)c * npix] * a;
}

__global__ void k_conv1x1(const float* __restrict__ in, int Cin,
                          const float* __restrict__ w,
                          const float* __restrict__ bias,
                          float* __restrict__ out, int Cout, int HW) {
  int bc = blockIdx.x;
  int b = bc / Cout, co = bc % Cout;
  int pix = threadIdx.x;
  int npix = HW * HW;
  float acc = bias[co];
  const float* ip = in + (size_t)b * Cin * npix + pix;
  const float* wp = w + (size_t)co * Cin;
  for (int c = 0; c < Cin; ++c) acc += wp[c] * ip[(size_t)c * npix];
  out[((size_t)b * Cout + co) * npix + pix] = acc;
}

// ---------------- 4) weight convert+transpose f32[K,M] -> bf16[M,K] ----------------
__global__ void k_prep_w(const float* __restrict__ w, unsigned short* __restrict__ wt) {
  int i = blockIdx.x * blockDim.x + threadIdx.x;   // over CD*CK
  int m = i / CK, k = i % CK;
  wt[(size_t)m * CK + k] = f2bf(w[(size_t)k * CD + m]);
}

// per-pair inputs: [ent(b,hi) || amap(b,:,hi,ti)] and [ent(b,ti) || amap(b,:,hi,ti)]
__global__ void k_build_pairs(const int* __restrict__ hts,
                              const float* __restrict__ ent,
                              const float* __restrict__ amap,   // NCHW [B,256,32,32]
                              unsigned short* __restrict__ inH,
                              unsigned short* __restrict__ inT) {
  int n = blockIdx.x;
  int b = n / CP;
  int hi = hts[n * 2], ti = hts[n * 2 + 1];
  const float* eh = ent + ((size_t)b * CNE + hi) * CD;
  const float* et = ent + ((size_t)b * CNE + ti) * CD;
  unsigned short* oh = inH + (size_t)n * CK;
  unsigned short* ot = inT + (size_t)n * CK;
  for (int i = threadIdx.x; i < CK; i += blockDim.x) {
    float hv, tv;
    if (i < CD) { hv = eh[i]; tv = et[i]; }
    else {
      int ch = i - CD;
      float av = amap[(((size_t)b * COC + ch) * CNE + hi) * CNE + ti];
      hv = av; tv = av;
    }
    oh[i] = f2bf(hv); ot[i] = f2bf(tv);
  }
}

// ---------------- 5) WMMA GEMM: tanh(A[3968,1024] x Bt^T + bias) ----------------
// A tile (16 x 1024 bf16, 32KB) is DMA'd into LDS once per block via the async
// global->LDS path (ASYNCcnt, inline asm), then each of the 4 waves computes a
// 16x64 output slab: one LDS A-fragment feeds 4 WMMAs against 4 global B-frags.
static constexpr int LDS_AST = 1032;   // bf16 units per LDS A row = 2064B (16B-aligned, bank-skewed)

__global__ __launch_bounds__(128) void k_gemm_tanh(
    const unsigned short* __restrict__ A,    // [CNPAIR, CK] bf16 bits
    const unsigned short* __restrict__ Bt,   // [CD, CK]     bf16 bits (feature-major)
    const float* __restrict__ bias,          // [CD]
    float* __restrict__ out) {               // [CNPAIR, CD] f32
  __shared__ unsigned short sA[16 * LDS_AST];
  int tid  = threadIdx.x;
  int wave = tid >> 5;
  int lane = tid & 31;
  int half = lane >> 4;
  int ln   = lane & 15;
  int rowBase  = blockIdx.x * 16;                 // 248 row tiles
  int featBase = blockIdx.y * 256 + wave * 64;    // 3 * 4 waves * 64 features

  // stage A tile: 2048 x b128 async DMA transfers, 16 per thread (uniform trip count)
  for (int c = tid; c < 2048; c += 128) {
    int r  = c >> 7;           // row 0..15
    int kc = c & 127;          // 8-bf16 chunk within row
    const unsigned short* g = A + (size_t)(rowBase + r) * CK + kc * 8;
    unsigned short* l = &sA[r * LDS_AST + kc * 8];
    unsigned long long ga = (unsigned long long)g;                 // 64-bit global addr
    unsigned la = (unsigned)(unsigned long long)(lds_us*)l;        // LDS byte offset (as3)
    asm volatile("global_load_async_to_lds_b128 %0, %1, off"
                 :: "v"(la), "v"(ga) : "memory");
  }
#if __has_builtin(__builtin_amdgcn_s_wait_asynccnt)
  __builtin_amdgcn_s_wait_asynccnt(0);
#else
  asm volatile("s_wait_asynccnt 0x0" ::: "memory");
#endif
  __syncthreads();

  const unsigned short* bp0 = Bt + (size_t)(featBase + ln) * CK;
  const unsigned short* bp1 = bp0 + (size_t)16 * CK;
  const unsigned short* bp2 = bp0 + (size_t)32 * CK;
  const unsigned short* bp3 = bp0 + (size_t)48 * CK;
  const unsigned short* la  = &sA[ln * LDS_AST];

  v8f acc0 = {}, acc1 = {}, acc2 = {}, acc3 = {};
#pragma unroll 4
  for (int k0 = 0; k0 < CK; k0 += 32) {
    // A frag, contiguous k order used here: lanes 0-15 K=0..15, lanes 16-31 K=16..31 (matches B)
    FragBF fa;
    fa.u[0] = *(const uint4*)(la + k0 + 16 * half);        // ds_load_b128
    fa.u[1] = *(const uint4*)(la + k0 + 16 * half + 8);
    // B frag, same contiguous k order as A: lanes 0-15 K=0..15, lanes 16-31 K=16..31 (col = lane%16)
    FragBF f0, f1, f2, f3;
    f0.u[0] = *(const uint4*)(bp0 + k0 + 16 * half);
    f0.u[1] = *(const uint4*)(bp0 + k0 + 16 * half + 8);
    f1.u[0] = *(const uint4*)(bp1 + k0 + 16 * half);
    f1.u[1] = *(const uint4*)(bp1 + k0 + 16 * half + 8);
    f2.u[0] = *(const uint4*)(bp2 + k0 + 16 * half);
    f2.u[1] = *(const uint4*)(bp2 + k0 + 16 * half + 8);
    f3.u[0] = *(const uint4*)(bp3 + k0 + 16 * half);
    f3.u[1] = *(const uint4*)(bp3 + k0 + 16 * half + 8);
    __builtin_prefetch((const void*)(bp0 + k0 + 128), 0, 3);   // global_prefetch_b8
    __builtin_prefetch((const void*)(bp2 + k0 + 128), 0, 3);
    acc0 = __builtin_amdgcn_wmma_f32_16x16x32_bf16(false, fa.v, false, f0.v, (short)0, acc0, false, false);
    acc1 = __builtin_amdgcn_wmma_f32_16x16x32_bf16(false, fa.v, false, f1.v, (short)0, acc1, false, false);
    acc2 = __builtin_amdgcn_wmma_f32_16x16x32_bf16(false, fa.v, false, f2.v, (short)0, acc2, false, false);
    acc3 = __builtin_amdgcn_wmma_f32_16x16x32_bf16(false, fa.v, false, f3.v, (short)0, acc3, false, false);
  }

  // D layout: VGPR r -> M = r + 8*half, N = lane%16
  v8f accs[4] = {acc0, acc1, acc2, acc3};
#pragma unroll
  for (int t = 0; t < 4; ++t) {
    int feat = featBase + 16 * t + ln;
    float bv = bias[feat];
#pragma unroll
    for (int r = 0; r < 8; ++r) {
      int row = rowBase + r + 8 * half;
      out[(size_t)row * CD + feat] = tanhf(accs[t][r] + bv);
    }
  }
}

// ---------------- 6) fused bilinear + decoder ----------------
// out[n,c] = sum_k hs[n,k*64+i] * W[k,i,j,c] * ts[n,k*64+j] + db[c]
__global__ void k_decoder(const float* __restrict__ hs,
                          const float* __restrict__ ts,
                          const float* __restrict__ W,    // [49152, 2]
                          const float* __restrict__ db,
                          float* __restrict__ out) {
  int n = blockIdx.x;
  int tid = threadIdx.x;          // 128 threads
  __shared__ float hss[CD];
  __shared__ float tss[CD];
  for (int i = tid; i < CD; i += blockDim.x) {
    hss[i] = hs[(size_t)n * CD + i];
    tss[i] = ts[(size_t)n * CD + i];
  }
  __syncthreads();
  float a0 = 0.f, a1 = 0.f;
  for (int row = tid; row < CD; row += blockDim.x) {   // row = k*64 + i
    const float* wr = W + (size_t)row * 128;           // W[(row*64+j)*2 + c]
    const float* tb = tss + (row & ~63);
    float s0 = 0.f, s1 = 0.f;
    for (int j = 0; j < 64; ++j) {
      float tv = tb[j];
      s0 += wr[j * 2]     * tv;
      s1 += wr[j * 2 + 1] * tv;
    }
    float h = hss[row];
    a0 += h * s0; a1 += h * s1;
  }
  __shared__ float r0[128], r1[128];
  r0[tid] = a0; r1[tid] = a1;
  __syncthreads();
  for (int s = 64; s > 0; s >>= 1) {
    if (tid < s) { r0[tid] += r0[tid + s]; r1[tid] += r1[tid + s]; }
    __syncthreads();
  }
  if (tid == 0) {
    out[(size_t)n * 2]     = r0[0] + db[0];
    out[(size_t)n * 2 + 1] = r1[0] + db[1];
  }
}

// ---------------- launcher ----------------
extern "C" void kernel_launch(void* const* d_in, const int* in_sizes, int n_in,
                              void* d_out, int out_size, void* d_ws, size_t ws_size,
                              hipStream_t stream) {
  (void)in_sizes; (void)n_in; (void)out_size; (void)ws_size;
  const float* x       = (const float*)d_in[0];
  // d_in[1] attention_mask: unused by the reference computation (only shape)
  const int*   epos    = (const int*)d_in[2];
  const int*   hts     = (const int*)d_in[3];
  const float* enc1_w  = (const float*)d_in[4];
  const float* enc1_b  = (const float*)d_in[5];
  const float* enc2_w  = (const float*)d_in[6];
  const float* enc2_b  = (const float*)d_in[7];
  const float* bott_w  = (const float*)d_in[8];
  const float* bott_b  = (const float*)d_in[9];
  const float* ag2_wg  = (const float*)d_in[10];
  const float* ag2_wx  = (const float*)d_in[11];
  const float* ag2_psi = (const float*)d_in[12];
  const float* dec2_w  = (const float*)d_in[13];
  const float* dec2_b  = (const float*)d_in[14];
  const float* ag1_wg  = (const float*)d_in[15];
  const float* ag1_wx  = (const float*)d_in[16];
  const float* ag1_psi = (const float*)d_in[17];
  const float* dec1_w  = (const float*)d_in[18];
  const float* dec1_b  = (const float*)d_in[19];
  const float* fin_w   = (const float*)d_in[20];
  const float* fin_b   = (const float*)d_in[21];
  const float* head_w  = (const float*)d_in[22];
  const float* head_b  = (const float*)d_in[23];
  const float* tail_w  = (const float*)d_in[24];
  const float* tail_b  = (const float*)d_in[25];
  const float* dec_w   = (const float*)d_in[26];
  const float* dec_b   = (const float*)d_in[27];
  float* out = (float*)d_out;

  char* base = (char*)d_ws;
  size_t off = 0;
  auto alloc = [&](size_t bytes) -> void* {
    void* p = base + off;
    off = (off + bytes + 255) & ~(size_t)255;
    return p;
  };
  float* ent  = (float*)alloc((size_t)CB * CNE * CD * 4);
  float* nrm  = (float*)alloc((size_t)CB * CNE * CD * 4);
  float* cosm = (float*)alloc((size_t)CB * CNE * CNE * 4);
  float* c1   = (float*)alloc((size_t)CB * 64  * 1024 * 4);
  float* p1   = (float*)alloc((size_t)CB * 64  * 256  * 4);
  float* c2   = (float*)alloc((size_t)CB * 128 * 256  * 4);
  float* p2   = (float*)alloc((size_t)CB * 128 * 64   * 4);
  float* c3   = (float*)alloc((size_t)CB * 256 * 64   * 4);
  float* u2   = (float*)alloc((size_t)CB * 256 * 256  * 4);
  float* t2   = (float*)alloc((size_t)CB * 128 * 256  * 4);
  float* g2   = (float*)alloc((size_t)CB * 128 * 256  * 4);
  float* d2   = (float*)alloc((size_t)CB * 128 * 256  * 4);
  float* u1   = (float*)alloc((size_t)CB * 128 * 1024 * 4);
  float* t1   = (float*)alloc((size_t)CB * 64  * 1024 * 4);
  float* g1   = (float*)alloc((size_t)CB * 64  * 1024 * 4);
  float* d1   = (float*)alloc((size_t)CB * 64  * 1024 * 4);
  float* amap = (float*)alloc((size_t)CB * COC * 1024 * 4);
  unsigned short* headwt = (unsigned short*)alloc((size_t)CD * CK * 2);
  unsigned short* tailwt = (unsigned short*)alloc((size_t)CD * CK * 2);
  unsigned short* inH    = (unsigned short*)alloc((size_t)CNPAIR * CK * 2);
  unsigned short* inT    = (unsigned short*)alloc((size_t)CNPAIR * CK * 2);
  float* hsb = (float*)alloc((size_t)CNPAIR * CD * 4);
  float* tsb = (float*)alloc((size_t)CNPAIR * CD * 4);

  // entities + cosine map
  k_gather_norm<<<CB * CNE, 256, 0, stream>>>(x, epos, ent, nrm);
  k_cos<<<CB, dim3(32, 32), 0, stream>>>(nrm, cosm);

  // UNet
  k_conv3x3<<<CB * 64, 1024, 0, stream>>>(cosm, 1, nullptr, 0, enc1_w, enc1_b, c1, 64, 32);
  k_pool<<<CB * 64, 256, 0, stream>>>(c1, p1, 32);
  k_conv3x3<<<CB * 128, 256, 0, stream>>>(p1, 64, nullptr, 0, enc2_w, enc2_b, c2, 128, 16);
  k_pool<<<CB * 128, 64, 0, stream>>>(c2, p2, 16);
  k_conv3x3<<<CB * 256, 64, 0, stream>>>(p2, 128, nullptr, 0, bott_w, bott_b, c3, 256, 8);
  k_up2<<<CB * 256, 256, 0, stream>>>(c3, u2, 16);
  k_gate_t<<<CB * 128, 256, 0, stream>>>(u2, 256, c2, 128, ag2_wg, ag2_wx, t2, 128, 16);
  k_gate_apply<<<CB, 256, 0, stream>>>(t2, 128, ag2_psi, c2, g2, 16);
  k_conv3x3<<<CB * 128, 256, 0, stream>>>(u2, 256, g2, 128, dec2_w, dec2_b, d2, 128, 16);
  k_up2<<<CB * 128, 1024, 0, stream>>>(d2, u1, 32);
  k_gate_t<<<CB * 64, 1024, 0, stream>>>(u1, 128, c1, 64, ag1_wg, ag1_wx, t1, 64, 32);
  k_gate_apply<<<CB, 1024, 0, stream>>>(t1, 64, ag1_psi, c1, g1, 32);
  k_conv3x3<<<CB * 64, 1024, 0, stream>>>(u1, 128, g1, 64, dec1_w, dec1_b, d1, 64, 32);
  k_conv1x1<<<CB * COC, 1024, 0, stream>>>(d1, 64, fin_w, fin_b, amap, COC, 32);

  // head/tail GEMMs via bf16 WMMA (async A-tile staging to LDS)
  k_prep_w<<<(CD * CK) / 256, 256, 0, stream>>>(head_w, headwt);
  k_prep_w<<<(CD * CK) / 256, 256, 0, stream>>>(tail_w, tailwt);
  k_build_pairs<<<CNPAIR, 256, 0, stream>>>(hts, ent, amap, inH, inT);
  dim3 ggrid(CNPAIR / 16, CD / 256);   // (248, 3), 4 waves/block, 16x64 tile per wave
  k_gemm_tanh<<<ggrid, 128, 0, stream>>>(inH, headwt, head_b, hsb);
  k_gemm_tanh<<<ggrid, 128, 0, stream>>>(inT, tailwt, tail_b, tsb);

  // fused bilinear + decoder
  k_decoder<<<CNPAIR, 128, 0, stream>>>(hsb, tsb, dec_w, dec_b, out);
}
